// Attention_57011395887690
// MI455X (gfx1250) — matrix operand order repaired and checked by reading, not compiled, not run on hardware
//
#include <hip/hip_runtime.h>

// ---------------------------------------------------------------------------
// Attention (B=4, S=2048, D=1024), fp32 in/out, bf16 WMMA compute on gfx1250.
//
// scores = (x@Wq^T) @ view(x@Wk^T as [D,S]) / D ; softmax ; @ (x@Wv^T)
// Every matmul is a plain row-major GEMM (the reference's reshape is a flat
// reinterpretation; W is read transposed for the projections).
//
// GEMM: 128x128 block tile, BK=32, 8 waves (2x4), wave tile 64x32 (8 accs),
// double-buffered LDS. A-tiles for bf16 GEMMs are staged by the Tensor Data
// Mover (tensor_load_to_lds + s_wait_tensorcnt); B-tiles are staged manually
// (K-transposed) with global_prefetch_b8 streaming the tile after next.
// ---------------------------------------------------------------------------

typedef __bf16 v16bf __attribute__((ext_vector_type(16)));
typedef float  v8f   __attribute__((ext_vector_type(8)));
typedef unsigned u32x4 __attribute__((ext_vector_type(4)));
typedef int      i32x8 __attribute__((ext_vector_type(8)));
typedef int      i32x4 __attribute__((ext_vector_type(4)));

constexpr int kB = 4;
constexpr int kS = 2048;
constexpr int kD = 1024;

__device__ __forceinline__ unsigned short f2bf(float f) {
  union { float f; unsigned u; } v; v.f = f;
  unsigned u = v.u;
  u += 0x7FFFu + ((u >> 16) & 1u);   // round-to-nearest-even
  return (unsigned short)(u >> 16);
}

__device__ __forceinline__ unsigned pack2bf(float a, float b) {
  return (unsigned)f2bf(a) | ((unsigned)f2bf(b) << 16);
}

union Frag { v16bf v; uint4 q[2]; };

// AMODE: 1 = A is fp32 row-major (convert to bf16 while staging, manual path)
//        0 = A is bf16 row-major (staged by TDM tensor_load_to_lds)
// BMODE: 1 = logical B[k][n] = W[n*ldb + k], W fp32 [N,K] row-major
//        0 = logical B[k][n] = Bh[k*ldb + n], bf16 row-major
// OUTBF: 1 = store bf16, 0 = store f32 (scaled)
template <int AMODE, int BMODE, int OUTBF>
__global__ __launch_bounds__(256)
void gemm_wmma_bf16(const void* __restrict__ Ap, const void* __restrict__ Bp,
                    void* __restrict__ Cp, int M, int N, int K,
                    int lda, int ldb, int ldc, float scale,
                    long long strideA, long long strideB, long long strideC)
{
  constexpr int BM = 128, BN = 128, BK = 32;
  __shared__ __align__(16) unsigned short lds_a[2][BM * BK];  // [m][k] 2x8KB
  __shared__ __align__(16) unsigned short lds_b[2][BN * BK];  // [n][k] 2x8KB

  const int tid   = threadIdx.x;
  const int lane  = tid & 31;
  const int wid   = tid >> 5;        // 8 waves
  const int waveM = wid >> 2;        // 2 wave-rows of 64
  const int waveN = wid & 3;         // 4 wave-cols of 32
  const int lrow  = lane & 15;
  const int lhalf = lane >> 4;

  const int m0 = blockIdx.y * BM;
  const int n0 = blockIdx.x * BN;
  const int batch = blockIdx.z;

  const float*          Af = nullptr;
  const unsigned short* Ah = nullptr;
  if constexpr (AMODE == 1) Af = (const float*)Ap + (long long)batch * strideA;
  else                      Ah = (const unsigned short*)Ap + (long long)batch * strideA;
  const float*          Bf = nullptr;
  const unsigned short* Bh = nullptr;
  if constexpr (BMODE == 1) Bf = (const float*)Bp + (long long)batch * strideB;
  else                      Bh = (const unsigned short*)Bp + (long long)batch * strideB;

  v8f acc[4][2];
#pragma unroll
  for (int i = 0; i < 4; ++i)
#pragma unroll
    for (int j = 0; j < 2; ++j) acc[i][j] = v8f{0, 0, 0, 0, 0, 0, 0, 0};

  // ---- staging thread mapping -------------------------------------------
  const int sr  = tid >> 3;          // 0..31
  const int sc  = (tid & 7) * 4;     // 0,4,...,28
  const int bk2 = (tid >> 4) * 2;    // even k: 0,2,...,30
  const int bng = (tid & 15) * 8;    // n group: 0,8,...,120

  float4 raf[4];                     // A prefetch registers (fp32 path)
  float4 rbf[4]; uint4 rbh[2];       // B prefetch registers

  // --- TDM: DMA a [BM x BK] bf16 tile (row-major, stride lda) into LDS ---
  auto tdmLoadA = [&](int kb, int buf) {
    const unsigned long long ga =
        (unsigned long long)(Ah + (size_t)m0 * lda + kb);
    const unsigned la = (unsigned)(unsigned long long)(&lds_a[buf][0]);
    u32x4 g0;
    g0[0] = 1u;                                  // count=1, user descriptor
    g0[1] = la;                                  // lds_addr (bytes)
    g0[2] = (unsigned)(ga & 0xFFFFFFFFu);        // global_addr[31:0]
    g0[3] = (unsigned)((ga >> 32) & 0x01FFFFFFu) // global_addr[56:32]
            | 0x80000000u;                       // type=2 (bits 127:126 = 10b)
    i32x8 g1;
    g1[0] = (int)(1u << 16);                     // data_size=2B; no multicast
    g1[1] = (int)(((unsigned)lda & 0xFFFFu) << 16);            // dim0[15:0]
    g1[2] = (int)((((unsigned)lda >> 16) & 0xFFFFu)
            | (((unsigned)M & 0xFFFFu) << 16));  // dim0[31:16] | dim1[15:0]
    g1[3] = (int)((((unsigned)M >> 16) & 0xFFFFu)
            | ((unsigned)BK << 16));             // dim1[31:16] | tile_dim0
    g1[4] = (int)(unsigned)BM;                   // tile_dim1 (tile_dim2 = 0)
    g1[5] = (int)(unsigned)lda;                  // dim0_stride[31:0]
    g1[6] = 0;                                   // dim0_stride[47:32] | d1s lo
    g1[7] = 0;
    const i32x4 z4 = {0, 0, 0, 0};
    const i32x8 z8 = {0, 0, 0, 0, 0, 0, 0, 0};
    __builtin_amdgcn_tensor_load_to_lds(g0, g1, z4, z4, z8, 0);
  };

  auto loadA = [&](int kb) {                     // fp32 manual path only
#pragma unroll
    for (int p = 0; p < 4; ++p) {
      const int row = p * 32 + sr;
      raf[p] = *(const float4*)(Af + (size_t)(m0 + row) * lda + kb + sc);
    }
  };
  auto storeA = [&](int buf) {
#pragma unroll
    for (int p = 0; p < 4; ++p) {
      const int row = p * 32 + sr;
      uint2 u;
      u.x = pack2bf(raf[p].x, raf[p].y);
      u.y = pack2bf(raf[p].z, raf[p].w);
      *(uint2*)&lds_a[buf][row * BK + sc] = u;
    }
  };
  auto loadB = [&](int kb) {
    if constexpr (BMODE == 1) {
#pragma unroll
      for (int p = 0; p < 4; ++p) {
        const int n = p * 32 + sr;
        rbf[p] = *(const float4*)(Bf + (size_t)(n0 + n) * ldb + kb + sc);
      }
    } else {
      rbh[0] = *(const uint4*)(Bh + (size_t)(kb + bk2)     * ldb + n0 + bng);
      rbh[1] = *(const uint4*)(Bh + (size_t)(kb + bk2 + 1) * ldb + n0 + bng);
    }
  };
  auto storeB = [&](int buf) {
    if constexpr (BMODE == 1) {
#pragma unroll
      for (int p = 0; p < 4; ++p) {
        const int n = p * 32 + sr;
        uint2 u;
        u.x = pack2bf(rbf[p].x, rbf[p].y);
        u.y = pack2bf(rbf[p].z, rbf[p].w);
        *(uint2*)&lds_b[buf][n * BK + sc] = u;
      }
    } else {
      // transpose in-register: pack (k, k+1) pairs -> one b32 store per n
      const unsigned short* s0 = (const unsigned short*)&rbh[0];
      const unsigned short* s1 = (const unsigned short*)&rbh[1];
#pragma unroll
      for (int j = 0; j < 8; ++j) {
        const unsigned u = (unsigned)s0[j] | ((unsigned)s1[j] << 16);
        *(unsigned*)&lds_b[buf][(bng + j) * BK + bk2] = u;
      }
    }
  };
  auto prefetchFar = [&](int kb) {   // stream tile-after-next into L2
    if constexpr (AMODE == 1)
      __builtin_prefetch(Af + (size_t)(m0 + sr) * lda + kb + sc, 0, 3);
    if constexpr (BMODE == 1)
      __builtin_prefetch(Bf + (size_t)(n0 + sr) * ldb + kb + sc, 0, 3);
    else
      __builtin_prefetch(Bh + (size_t)(kb + bk2) * ldb + n0 + bng, 0, 3);
  };

  auto compute = [&](int buf) {
    // Preload all six fragments, then burst 8 back-to-back WMMAs.
    Frag b0, b1, a[4];
    const unsigned short* pb = &lds_b[buf][(waveN * 32 + lrow) * BK + lhalf * 16];
    b0.q[0] = *(const uint4*)pb;       b0.q[1] = *(const uint4*)(pb + 8);
    const unsigned short* pb1 = pb + 16 * BK;
    b1.q[0] = *(const uint4*)pb1;      b1.q[1] = *(const uint4*)(pb1 + 8);
#pragma unroll
    for (int fm = 0; fm < 4; ++fm) {
      const unsigned short* pa =
          &lds_a[buf][(waveM * 64 + fm * 16 + lrow) * BK + lhalf * 16];
      a[fm].q[0] = *(const uint4*)pa;  a[fm].q[1] = *(const uint4*)(pa + 8);
    }
#pragma unroll
    for (int fm = 0; fm < 4; ++fm) {
      acc[fm][0] = __builtin_amdgcn_wmma_f32_16x16x32_bf16(
          false, a[fm].v, false, b0.v, (short)0, acc[fm][0], false, false);
      acc[fm][1] = __builtin_amdgcn_wmma_f32_16x16x32_bf16(
          false, a[fm].v, false, b1.v, (short)0, acc[fm][1], false, false);
    }
  };

  // ---- pipelined main loop ----------------------------------------------
  if constexpr (AMODE == 1) { loadA(0); storeA(0); }
  else if (wid == 0) tdmLoadA(0, 0);             // wave-uniform branch
  loadB(0); storeB(0);
  if constexpr (AMODE == 0) {
    if (wid == 0) __builtin_amdgcn_s_wait_tensorcnt(0);
  }
  __syncthreads();

  int cur = 0;
  for (int kb = 0; kb < K; kb += BK) {
    const bool hasNext = (kb + BK) < K;          // wave-uniform
    if (hasNext) {
      if constexpr (AMODE == 1) loadA(kb + BK);  // -> regs, overlaps compute
      else if (wid == 0) tdmLoadA(kb + BK, cur ^ 1);  // DMA, overlaps compute
      loadB(kb + BK);
    }
    if (kb + 2 * BK < K) prefetchFar(kb + 2 * BK);
    compute(cur);
    __syncthreads();                             // all reads of lds[cur] done
    if (hasNext) {
      if constexpr (AMODE == 1) storeA(cur ^ 1);
      storeB(cur ^ 1);
      if constexpr (AMODE == 0) {
        if (wid == 0) __builtin_amdgcn_s_wait_tensorcnt(0);
      }
    }
    __syncthreads();                             // lds[cur^1] fully staged
    cur ^= 1;
  }

  // ---- epilogue: C/D f32 16x16 layout: col = lane&15, row = (lane>>4)*8+r
  float*          Cf = nullptr;
  unsigned short* Ch = nullptr;
  if constexpr (OUTBF) Ch = (unsigned short*)Cp + (long long)batch * strideC;
  else                 Cf = (float*)Cp + (long long)batch * strideC;

  const int colBase = n0 + waveN * 32 + (lane & 15);
  const int rowBase = m0 + waveM * 64 + (lane >> 4) * 8;
#pragma unroll
  for (int fm = 0; fm < 4; ++fm)
#pragma unroll
    for (int fn = 0; fn < 2; ++fn) {
      const int col = colBase + fn * 16;
#pragma unroll
      for (int r = 0; r < 8; ++r) {
        const int row = rowBase + fm * 16 + r;
        const float val = acc[fm][fn][r] * scale;
        if constexpr (OUTBF) Ch[(size_t)row * ldc + col] = f2bf(val);
        else                 Cf[(size_t)row * ldc + col] = val;
      }
    }
}

// Row softmax over length 2048 (one 256-thread block per row), bf16 output.
__global__ __launch_bounds__(256)
void softmax_rows_bf16(const float* __restrict__ S, unsigned short* __restrict__ O)
{
  const int row = blockIdx.x;
  const float* r = S + (size_t)row * kS;
  unsigned short* o = O + (size_t)row * kS;
  const int tid = threadIdx.x;
  __shared__ float red[256];

  float v[8];
  float m = -3.4e38f;
#pragma unroll
  for (int j = 0; j < 8; ++j) { v[j] = r[tid + j * 256]; m = fmaxf(m, v[j]); }
  red[tid] = m; __syncthreads();
  for (int s = 128; s > 0; s >>= 1) {
    if (tid < s) red[tid] = fmaxf(red[tid], red[tid + s]);
    __syncthreads();
  }
  m = red[0]; __syncthreads();

  float sum = 0.f;
#pragma unroll
  for (int j = 0; j < 8; ++j) { v[j] = __expf(v[j] - m); sum += v[j]; }
  red[tid] = sum; __syncthreads();
  for (int s = 128; s > 0; s >>= 1) {
    if (tid < s) red[tid] += red[tid + s];
    __syncthreads();
  }
  const float inv = 1.f / red[0];
#pragma unroll
  for (int j = 0; j < 8; ++j) o[tid + j * 256] = f2bf(v[j] * inv);
}

extern "C" void kernel_launch(void* const* d_in, const int* in_sizes, int n_in,
                              void* d_out, int out_size, void* d_ws, size_t ws_size,
                              hipStream_t stream)
{
  const float* x  = (const float*)d_in[0];
  const float* WQ = (const float*)d_in[1];
  const float* WK = (const float*)d_in[2];
  const float* WV = (const float*)d_in[3];
  float* out = (float*)d_out;

  constexpr long long SD  = (long long)kS * kD;        // 2,097,152
  constexpr long long SS  = (long long)kS * kS;        // 4,194,304
  constexpr long long BSD = (long long)kB * SD;        // 8,388,608

  // Workspace layout (144 MB total):
  //  [0)        Qb  bf16 [B*S, D]   16 MB
  //  [16 MB)    Kb  bf16 [B, S*D]   16 MB  (viewed [D,S] per batch for scores)
  //  [32 MB)    Vb  bf16 [B, S*D]   16 MB
  //  [48 MB)    Sco f32  [B, S, S]  64 MB
  //  [112 MB)   Ab  bf16 [B, S, S]  32 MB
  char* ws = (char*)d_ws;
  unsigned short* Qb  = (unsigned short*)ws;
  unsigned short* Kb  = Qb + BSD;
  unsigned short* Vb  = Kb + BSD;
  float*          Sco = (float*)(ws + 3 * BSD * sizeof(unsigned short));
  unsigned short* Ab  = (unsigned short*)((char*)Sco + (long long)kB * SS * sizeof(float));

  const dim3 blk(256, 1, 1);

  // 1) Projections: [8192,1024] = x @ W^T  (fp32 in, bf16 out)
  {
    const dim3 grd(kD / 128, (kB * kS) / 128, 1);
    gemm_wmma_bf16<1, 1, 1><<<grd, blk, 0, stream>>>(x, WQ, Qb, kB * kS, kD, kD,
                                                     kD, kD, kD, 1.0f, 0, 0, 0);
    gemm_wmma_bf16<1, 1, 1><<<grd, blk, 0, stream>>>(x, WK, Kb, kB * kS, kD, kD,
                                                     kD, kD, kD, 1.0f, 0, 0, 0);
    gemm_wmma_bf16<1, 1, 1><<<grd, blk, 0, stream>>>(x, WV, Vb, kB * kS, kD, kD,
                                                     kD, kD, kD, 1.0f, 0, 0, 0);
  }
  // 2) scores = Q @ view(K,[D,S]) * (1/D)   (bf16 in, f32 out), batched, TDM-A
  {
    const dim3 grd(kS / 128, kS / 128, kB);
    gemm_wmma_bf16<0, 0, 0><<<grd, blk, 0, stream>>>(Qb, Kb, Sco, kS, kS, kD,
                                                     kD, kS, kS, 1.0f / (float)kD,
                                                     SD, SD, SS);
  }
  // 3) Row softmax -> bf16 attn
  softmax_rows_bf16<<<dim3(kB * kS, 1, 1), blk, 0, stream>>>(Sco, Ab);
  // 4) out = attn @ V   (bf16 in, f32 out), batched, TDM-A
  {
    const dim3 grd(kD / 128, kS / 128, kB);
    gemm_wmma_bf16<0, 0, 0><<<grd, blk, 0, stream>>>(Ab, Vb, out, kS, kD, kS,
                                                     kS, kD, kD, 1.0f,
                                                     SS, SD, SD);
  }
}
